// LSTMModel_21973052686520
// MI455X (gfx1250) — matrix operand order repaired and checked by reading, not compiled, not run on hardware
//
#include <hip/hip_runtime.h>
#include <hip/hip_bf16.h>

typedef __attribute__((ext_vector_type(16))) __bf16 v16bf;
typedef __attribute__((ext_vector_type(8)))  __bf16 v8bf;
typedef __attribute__((ext_vector_type(8)))  float  v8f;

#define BATCH 64
#define SEQ   256
#define DIN   512
#define HID   1024
#define NGATE 4096   // 4*HID
#define NBLK  64     // persistent layer kernel grid size

// LDS layout of persistent layer kernel (bytes):
//   [0,      131072) : packed Whh B-tiles, 4 gates x 32KB (staged once, async)
//   [131072, 147456) : gate exchange, 4*64*16 fp32
//   [147456, 151552) : cell state,   64*16 fp32
#define LDS_B_BYTES   (4 * 32768)
#define LDS_G_OFF     LDS_B_BYTES
#define LDS_C_OFF     (LDS_G_OFF + 4 * 64 * 16 * 4)
#define LDS_TOTAL     (LDS_C_OFF + 64 * 16 * 4)

__device__ inline float sigmoidf_(float x) {
  return 1.0f / (1.0f + __expf(-x));
}

// ---------------------------------------------------------------------------
// Pack fp32 weight W[N=4H rows][K cols] into bf16, pre-swizzled into WMMA
// B-fragment layout: per 32x16 (KxN) tile, 32 lanes x 16 bf16 contiguous.
// lane = (k_local>>4)*16 + n_local ; element = k_local & 15
// ---------------------------------------------------------------------------
__global__ void pack_b_kernel(const float* __restrict__ W,
                              __hip_bfloat16* __restrict__ P,
                              int N, int K) {
  size_t idx = (size_t)blockIdx.x * blockDim.x + threadIdx.x;
  size_t total = (size_t)N * K;
  if (idx >= total) return;
  int n = (int)(idx / K);
  int k = (int)(idx % K);
  int kt = k >> 5, kl = k & 31;
  int half = kl >> 4;
  int nt = n >> 4, nl = n & 15;
  int lane = half * 16 + nl;
  size_t tile = (size_t)nt * (K >> 5) + kt;
  size_t dst = tile * 512 + (size_t)lane * 16 + (kl & 15);
  P[dst] = __float2bfloat16(W[idx]);
}

__global__ void f32_to_bf16_kernel(const float* __restrict__ in,
                                   __hip_bfloat16* __restrict__ out, size_t n) {
  size_t i = (size_t)blockIdx.x * blockDim.x + threadIdx.x;
  if (i < n) out[i] = __float2bfloat16(in[i]);
}

__global__ void zero_bf16_kernel(__hip_bfloat16* __restrict__ p, int n) {
  int i = blockIdx.x * blockDim.x + threadIdx.x;
  if (i < n) p[i] = __float2bfloat16(0.0f);
}

__global__ void zero_u32_kernel(unsigned int* __restrict__ p, int n) {
  int i = blockIdx.x * blockDim.x + threadIdx.x;
  if (i < n) p[i] = 0u;
}

// Assemble a 16-bit WMMA A-fragment for one lane from a bf16 row-major row.
// lanes 0-15 : VGPR0-7 = K 0..15
// lanes 16-31: VGPR0-7 = K 16..31
__device__ inline v16bf load_a_frag(const __bf16* __restrict__ Arow,
                                    int k0, int half) {
  v8bf a0 = *(const v8bf*)(Arow + k0 + half * 16);
  v8bf a1 = *(const v8bf*)(Arow + k0 + half * 16 + 8);
  return __builtin_shufflevector(a0, a1, 0, 1, 2, 3, 4, 5, 6, 7,
                                 8, 9, 10, 11, 12, 13, 14, 15);
}

__device__ inline v8f wmma_bf16(v16bf a, v16bf b, v8f c) {
  return __builtin_amdgcn_wmma_f32_16x16x32_bf16(
      false, a, false, b, (short)0, c, false, false);
}

// CDNA5 async global->LDS copy of 16 bytes per lane (ASYNCcnt-tracked).
// ldsAddr: LDS byte address (generic LDS ptr truncated to 32b = aperture low
// half); gaddr: 64-bit global address.
__device__ inline void async_copy16(unsigned ldsAddr, unsigned long long gaddr) {
  asm volatile("global_load_async_to_lds_b128 %0, %1, off"
               :: "v"(ldsAddr), "v"(gaddr) : "memory");
}
__device__ inline void async_wait0() {
  asm volatile("s_wait_asynccnt 0" ::: "memory");
}

// ---------------------------------------------------------------------------
// gx = A[M,K](bf16) @ packed-B(K x 4096) + bias1 + bias2 -> Out[M,4096] fp32
// block = 256 threads = 8 waves; block tile = 128 M x 64 N.
// 2x2 register blocking: wave (mp = w>>1, np = w&1) computes M tiles
// {2mp,2mp+1} x N tiles {2np,2np+1}: 2 A-frags x 2 B-frags -> 4 WMMA,
// halving operand bytes per WMMA vs 1x1. 4 column tiles staged in LDS.
// ---------------------------------------------------------------------------
__global__ void gemm_gx_kernel(const __hip_bfloat16* __restrict__ A,
                               const __hip_bfloat16* __restrict__ Bp,
                               const float* __restrict__ bias1,
                               const float* __restrict__ bias2,
                               float* __restrict__ Out,
                               int K) {
  extern __shared__ __align__(64) unsigned char lds_raw[];
  const int KT  = K >> 5;
  const int tid = threadIdx.x;
  const int ntBase = blockIdx.x * 4;     // 4 column tiles per block
  const int m0  = blockIdx.y * 128;      // row block base

  {
    const uint4* src = (const uint4*)(Bp + (size_t)ntBase * KT * 512);
    uint4* dst = (uint4*)lds_raw;
    const int nvec = 4 * KT * 64;        // 4 tiles * KT KB
    for (int i = tid; i < nvec; i += blockDim.x) dst[i] = src[i];
  }
  __syncthreads();

  const int wave = tid >> 5;
  const int lane = tid & 31;
  const int half = lane >> 4;
  const int r    = lane & 15;
  const int mp   = wave >> 1;            // 0..3
  const int np   = wave & 1;             // 0..1
  const __bf16* A0 =
      (const __bf16*)A + (size_t)(m0 + (mp * 2 + 0) * 16 + r) * K;
  const __bf16* A1 =
      (const __bf16*)A + (size_t)(m0 + (mp * 2 + 1) * 16 + r) * K;
  const unsigned char* B0 = lds_raw + (size_t)(np * 2 + 0) * KT * 1024;
  const unsigned char* B1 = lds_raw + (size_t)(np * 2 + 1) * KT * 1024;

  v8f acc00 = {}, acc01 = {}, acc10 = {}, acc11 = {};
  for (int kt = 0; kt < KT; ++kt) {
    const int k0 = kt << 5;
    v16bf a0 = load_a_frag(A0, k0, half);
    v16bf a1 = load_a_frag(A1, k0, half);
    v16bf b0 = *(const v16bf*)(B0 + kt * 1024 + lane * 32);
    v16bf b1 = *(const v16bf*)(B1 + kt * 1024 + lane * 32);
    acc00 = wmma_bf16(a0, b0, acc00);
    acc01 = wmma_bf16(a0, b1, acc01);
    acc10 = wmma_bf16(a1, b0, acc10);
    acc11 = wmma_bf16(a1, b1, acc11);
  }

  // D layout: VGPR i -> M = half*8 + i, N = lane&15
  const int nl = lane & 15;
  const int nA = (ntBase + np * 2 + 0) * 16 + nl;
  const int nB = (ntBase + np * 2 + 1) * 16 + nl;
  const float bsA = bias1[nA] + bias2[nA];
  const float bsB = bias1[nB] + bias2[nB];
#pragma unroll
  for (int i = 0; i < 8; ++i) {
    const int mA = m0 + (mp * 2 + 0) * 16 + half * 8 + i;
    const int mB = m0 + (mp * 2 + 1) * 16 + half * 8 + i;
    Out[(size_t)mA * NGATE + nA] = acc00[i] + bsA;
    Out[(size_t)mA * NGATE + nB] = acc01[i] + bsB;
    Out[(size_t)mB * NGATE + nA] = acc10[i] + bsA;
    Out[(size_t)mB * NGATE + nB] = acc11[i] + bsB;
  }
}

// ---------------------------------------------------------------------------
// Persistent LSTM layer kernel. grid = 64 blocks x 256 threads (8 waves).
// Block `blk` owns output columns [blk*16, blk*16+16) of ALL FOUR gates.
// Whh column tiles for the 4 gates (128KB) async-staged into LDS once and
// reused for all 256 timesteps. Per step:
//   wave w: gate = w&3, batch-tile pair = (w>>2)*2 + {0,1}
//   g = h_prev @ Whh^T + gx  -> LDS gate-exchange -> fused gate pointwise
//   h written (bf16) to global hbuf; device-wide release/acquire barrier.
// ---------------------------------------------------------------------------
__global__ void lstm_layer_kernel(const float* __restrict__ gx,
                                  const __hip_bfloat16* __restrict__ Whh_p,
                                  __hip_bfloat16* __restrict__ hbuf,
                                  __hip_bfloat16* __restrict__ HoutBf,
                                  float* __restrict__ OutF32,
                                  unsigned int* __restrict__ syncCtr) {
  extern __shared__ __align__(64) unsigned char lds[];
  float* gex = (float*)(lds + LDS_G_OFF);   // [4][64][16]
  float* cst = (float*)(lds + LDS_C_OFF);   // [64][16]

  const int tid = threadIdx.x;
  const int blk = blockIdx.x;               // 0..63 column group
  const int KT  = HID >> 5;                 // 32

  // Async-stage packed Whh B-tiles for the 4 gates (ASYNCcnt path).
#pragma unroll
  for (int g = 0; g < 4; ++g) {
    const int nt = g * (HID / 16) + blk;
    const unsigned char* src =
        (const unsigned char*)(Whh_p + (size_t)nt * KT * 512);
    for (int i = tid; i < KT * 64; i += blockDim.x) {
      async_copy16((unsigned)(size_t)(lds + g * 32768 + i * 16),
                   (unsigned long long)(size_t)(src + (size_t)i * 16));
    }
  }
  for (int i = tid; i < 64 * 16; i += blockDim.x) cst[i] = 0.0f;
  async_wait0();
  __syncthreads();

  const int wave = tid >> 5;
  const int lane = tid & 31;
  const int half = lane >> 4;
  const int r    = lane & 15;
  const int gw   = wave & 3;                // gate handled by this wave
  const int mh   = wave >> 2;               // batch half (tiles 2*mh, 2*mh+1)
  const unsigned char* ldsB = lds + gw * 32768;
  const int nl   = lane & 15;
  const int gcol = gw * HID + blk * 16 + nl;  // column in [0,4096)

  for (int t = 0; t < SEQ; ++t) {
    // Prefetch this step's gx operands behind the WMMA chain.
    __builtin_prefetch(&gx[((size_t)(mh * 32 + r) * SEQ + t) * NGATE + gcol],
                       0, 1);
    // ---- recurrent GEMM: two 16x16 batch tiles per wave, shared B frag ----
    v8f acc0 = {}, acc1 = {};
    const __bf16* A0 =
        (const __bf16*)hbuf + (size_t)((mh * 2 + 0) * 16 + r) * HID;
    const __bf16* A1 =
        (const __bf16*)hbuf + (size_t)((mh * 2 + 1) * 16 + r) * HID;
    for (int kt = 0; kt < KT; ++kt) {
      const int k0 = kt << 5;
      v16bf b  = *(const v16bf*)(ldsB + kt * 1024 + lane * 32);
      v16bf a0 = load_a_frag(A0, k0, half);
      v16bf a1 = load_a_frag(A1, k0, half);
      acc0 = wmma_bf16(a0, b, acc0);
      acc1 = wmma_bf16(a1, b, acc1);
    }
    // ---- D + gx -> LDS gate exchange ----
#pragma unroll
    for (int i = 0; i < 8; ++i) {
      const int m0 = mh * 32 + half * 8 + i;   // batch row of acc0 element
      const int m1 = m0 + 16;                  // batch row of acc1 element
      gex[(gw * 64 + m0) * 16 + nl] =
          acc0[i] + gx[((size_t)m0 * SEQ + t) * NGATE + gcol];
      gex[(gw * 64 + m1) * 16 + nl] =
          acc1[i] + gx[((size_t)m1 * SEQ + t) * NGATE + gcol];
    }
    __syncthreads();

    // ---- fused gate pointwise over 64 batches x 16 columns ----
    for (int p = tid; p < 64 * 16; p += blockDim.x) {
      const int b  = p >> 4;
      const int nn = p & 15;
      const float ig = sigmoidf_(gex[(0 * 64 + b) * 16 + nn]);
      const float fg = sigmoidf_(gex[(1 * 64 + b) * 16 + nn]);
      const float gg = tanhf(gex[(2 * 64 + b) * 16 + nn]);
      const float og = sigmoidf_(gex[(3 * 64 + b) * 16 + nn]);
      float c = fg * cst[p] + ig * gg;
      cst[p] = c;
      const float h = og * tanhf(c);
      const int n = blk * 16 + nn;
      hbuf[(size_t)b * HID + n] = __float2bfloat16(h);
      if (HoutBf) HoutBf[((size_t)b * SEQ + t) * HID + n] = __float2bfloat16(h);
      if (OutF32) {
        const float rl = h > 0.0f ? h : 0.0f;
        OutF32[(size_t)b * (SEQ * HID) + (size_t)t * HID + n] = sigmoidf_(rl);
      }
    }
    __syncthreads();

    // ---- device-wide barrier: publish hbuf, wait for all 64 blocks ----
    if (tid == 0) {
      __threadfence();  // release: flush hbuf stores device-scope
      __hip_atomic_fetch_add(syncCtr, 1u, __ATOMIC_RELEASE,
                             __HIP_MEMORY_SCOPE_AGENT);
      const unsigned target = (unsigned)(t + 1) * NBLK;
      while (__hip_atomic_load(syncCtr, __ATOMIC_ACQUIRE,
                               __HIP_MEMORY_SCOPE_AGENT) < target) {
        __builtin_amdgcn_s_sleep(2);
      }
    }
    __syncthreads();
    __threadfence();  // acquire side: invalidate stale cached hbuf lines
  }
}

// ---------------------------------------------------------------------------
extern "C" void kernel_launch(void* const* d_in, const int* in_sizes, int n_in,
                              void* d_out, int out_size, void* d_ws,
                              size_t ws_size, hipStream_t stream) {
  const float* x     = (const float*)d_in[0];
  const float* W_ih0 = (const float*)d_in[1];
  const float* W_hh0 = (const float*)d_in[2];
  const float* b_ih0 = (const float*)d_in[3];
  const float* b_hh0 = (const float*)d_in[4];
  const float* W_ih1 = (const float*)d_in[5];
  const float* W_hh1 = (const float*)d_in[6];
  const float* b_ih1 = (const float*)d_in[7];
  const float* b_hh1 = (const float*)d_in[8];
  float* out = (float*)d_out;

  // ---- workspace partition ----
  size_t off = 0;
  auto carve = [&](size_t bytes) -> void* {
    void* p = (char*)d_ws + off;
    off += (bytes + 255) & ~(size_t)255;
    return p;
  };
  __hip_bfloat16* pWih0 = (__hip_bfloat16*)carve((size_t)NGATE * DIN * 2);
  __hip_bfloat16* pWhh0 = (__hip_bfloat16*)carve((size_t)NGATE * HID * 2);
  __hip_bfloat16* pWih1 = (__hip_bfloat16*)carve((size_t)NGATE * HID * 2);
  __hip_bfloat16* pWhh1 = (__hip_bfloat16*)carve((size_t)NGATE * HID * 2);
  __hip_bfloat16* xbf   = (__hip_bfloat16*)carve((size_t)BATCH * SEQ * DIN * 2);
  __hip_bfloat16* h1bf  = (__hip_bfloat16*)carve((size_t)BATCH * SEQ * HID * 2);
  float* gx = (float*)carve((size_t)BATCH * SEQ * NGATE * 4);
  __hip_bfloat16* hprev = (__hip_bfloat16*)carve((size_t)BATCH * HID * 2);
  unsigned int* syncCtr = (unsigned int*)carve(2 * sizeof(unsigned int));

  const int M = BATCH * SEQ;  // 16384

  // Allow >64KB dynamic LDS (ignore errors; compile-only environment).
  static_assert(LDS_TOTAL == 151552, "lds layout");
  (void)hipFuncSetAttribute((const void*)lstm_layer_kernel,
                            hipFuncAttributeMaxDynamicSharedMemorySize,
                            LDS_TOTAL);
  (void)hipFuncSetAttribute((const void*)gemm_gx_kernel,
                            hipFuncAttributeMaxDynamicSharedMemorySize,
                            4 * (HID / 32) * 1024);

  // ---- pack weights into WMMA-B layout (bf16) ----
  {
    size_t n0 = (size_t)NGATE * DIN;
    pack_b_kernel<<<(unsigned)((n0 + 255) / 256), 256, 0, stream>>>(
        W_ih0, pWih0, NGATE, DIN);
    size_t n1 = (size_t)NGATE * HID;
    pack_b_kernel<<<(unsigned)((n1 + 255) / 256), 256, 0, stream>>>(
        W_hh0, pWhh0, NGATE, HID);
    pack_b_kernel<<<(unsigned)((n1 + 255) / 256), 256, 0, stream>>>(
        W_ih1, pWih1, NGATE, HID);
    pack_b_kernel<<<(unsigned)((n1 + 255) / 256), 256, 0, stream>>>(
        W_hh1, pWhh1, NGATE, HID);
  }

  // ---- convert x to bf16; reset sync counters ----
  {
    size_t nx = (size_t)BATCH * SEQ * DIN;
    f32_to_bf16_kernel<<<(unsigned)((nx + 255) / 256), 256, 0, stream>>>(
        x, xbf, nx);
  }
  zero_u32_kernel<<<1, 64, 0, stream>>>(syncCtr, 2);

  // ================= layer 0 =================
  {
    dim3 grid(NGATE / 64, M / 128);
    gemm_gx_kernel<<<grid, 256, 4 * (DIN / 32) * 1024, stream>>>(
        xbf, pWih0, b_ih0, b_hh0, gx, DIN);
  }
  zero_bf16_kernel<<<(BATCH * HID) / 256, 256, 0, stream>>>(hprev,
                                                            BATCH * HID);
  lstm_layer_kernel<<<NBLK, 256, LDS_TOTAL, stream>>>(
      gx, pWhh0, hprev, h1bf, nullptr, syncCtr + 0);

  // ================= layer 1 =================
  {
    dim3 grid(NGATE / 64, M / 128);
    gemm_gx_kernel<<<grid, 256, 4 * (HID / 32) * 1024, stream>>>(
        h1bf, pWih1, b_ih1, b_hh1, gx, HID);
  }
  zero_bf16_kernel<<<(BATCH * HID) / 256, 256, 0, stream>>>(hprev,
                                                            BATCH * HID);
  lstm_layer_kernel<<<NBLK, 256, LDS_TOTAL, stream>>>(
      gx, pWhh1, hprev, nullptr, out, syncCtr + 1);
}
